// RoleLabeler_11424613007966
// MI455X (gfx1250) — matrix operand order repaired and checked by reading, not compiled, not run on hardware
//
#include <hip/hip_runtime.h>

// ---------------------------------------------------------------------------
// Problem constants (match reference)
// ---------------------------------------------------------------------------
#define B_ 32
#define S_ 256
#define H_ 768
#define TWO_H 1536
#define FOUR_H 3072
#define HID_ 512
#define NSEL 20     // n_holders == n_targets == 20
#define NPAIR 400   // 20*20

typedef __attribute__((ext_vector_type(16))) __bf16 v16bf;
typedef __attribute__((ext_vector_type(8)))  float  v8f;

union Frag {
    v16bf v;
    uint4 u[2];
};

__device__ __forceinline__ unsigned short f2bf(float f) {
    unsigned int u = __float_as_uint(f);
    unsigned int r = u + 0x7FFFu + ((u >> 16) & 1u);   // round-to-nearest-even
    return (unsigned short)(r >> 16);
}

// ---------------------------------------------------------------------------
// Prep: fp32 -> bf16 copy (for X_word)
// ---------------------------------------------------------------------------
__global__ void f2bf_copy_kernel(const float* __restrict__ in,
                                 unsigned short* __restrict__ out, int n) {
    for (int i = blockIdx.x * blockDim.x + threadIdx.x; i < n;
         i += gridDim.x * blockDim.x)
        out[i] = f2bf(in[i]);
}

// Prep: fp32 [K][N] -> bf16 transposed [N][K] (for layer-1 weights)
__global__ void f2bf_transpose_kernel(const float* __restrict__ in,
                                      unsigned short* __restrict__ out,
                                      int K, int N) {
    int total = K * N;
    for (int i = blockIdx.x * blockDim.x + threadIdx.x; i < total;
         i += gridDim.x * blockDim.x) {
        int n = i / K;
        int k = i - n * K;
        out[(size_t)n * K + k] = f2bf(in[(size_t)k * N + n]);
    }
}

// ---------------------------------------------------------------------------
// Span scoring:  score[b, span] = relu(concat(x_s, x_e) @ W1 + b1) @ w2 + b2
// One block = 32 spans x 512 hidden cols. 8 waves x 4 col-tiles x 2 row-tiles.
// B fragment loaded once per (k-step, col-tile) and reused for both row tiles.
// ---------------------------------------------------------------------------
__global__ __launch_bounds__(256)
void span_score_kernel(const unsigned short* __restrict__ Xbf,   // [B][S][H] bf16
                       const int* __restrict__ starts,
                       const int* __restrict__ ends,
                       const unsigned short* __restrict__ W1t,   // [HID][2H] bf16
                       const float* __restrict__ b1,             // [HID]
                       const float* __restrict__ w2,             // [HID]
                       const float* __restrict__ b2,             // [1]
                       float* __restrict__ scores,               // [B][nspan]
                       int nspan) {
    const int b    = blockIdx.y;
    const int r0   = blockIdx.x * 32;
    const int tid  = threadIdx.x;
    const int wave = tid >> 5;
    const int lane = tid & 31;
    const int half = lane >> 4;
    const int mn   = lane & 15;

    // two row tiles: rows r0 + u*16 + mn
    const unsigned short* ps[2][2];
#pragma unroll
    for (int u = 0; u < 2; ++u) {
        int si = r0 + u * 16 + mn;
        if (si >= nspan) si = nspan - 1;
        const int srow = starts[si];
        const int erow = ends[si] - 1;
        ps[u][0] = Xbf + ((size_t)b * S_ + srow) * H_;
        ps[u][1] = Xbf + ((size_t)b * S_ + erow) * H_;
    }

    const int cw = wave * 64;
    const unsigned short* wb[4];
#pragma unroll
    for (int t = 0; t < 4; ++t)
        wb[t] = W1t + (size_t)(cw + t * 16 + mn) * TWO_H;

    v8f acc[2][4];
#pragma unroll
    for (int u = 0; u < 2; ++u)
#pragma unroll
        for (int t = 0; t < 4; ++t) acc[u][t] = (v8f){};

#pragma unroll
    for (int ch = 0; ch < 2; ++ch) {
        const unsigned short* base0 = ps[0][ch];
        const unsigned short* base1 = ps[1][ch];
        for (int off = 0; off < H_; off += 32) {
            const unsigned short* a0p = base0 + off + 8 * half;
            const unsigned short* a1p = base1 + off + 8 * half;
            Frag a0, a1;
            a0.u[0] = *(const uint4*)(a0p);
            a0.u[1] = *(const uint4*)(a0p + 16);
            a1.u[0] = *(const uint4*)(a1p);
            a1.u[1] = *(const uint4*)(a1p + 16);
            __builtin_prefetch((const void*)(a0p + 32), 0, 0);
            const int kg = ch * H_ + off + 8 * half;
#pragma unroll
            for (int t = 0; t < 4; ++t) {
                Frag w;
                const unsigned short* q = wb[t] + kg;
                w.u[0] = *(const uint4*)(q);
                w.u[1] = *(const uint4*)(q + 16);
                acc[0][t] = __builtin_amdgcn_wmma_f32_16x16x32_bf16(
                    false, a0.v, false, w.v, (short)0, acc[0][t], false, false);
                acc[1][t] = __builtin_amdgcn_wmma_f32_16x16x32_bf16(
                    false, a1.v, false, w.v, (short)0, acc[1][t], false, false);
            }
        }
    }

    // fused layer 2: relu(h + b1) . w2
    __shared__ float red[32];
    if (tid < 32) red[tid] = 0.f;
    __syncthreads();

#pragma unroll
    for (int u = 0; u < 2; ++u) {
        float sv[8];
#pragma unroll
        for (int v = 0; v < 8; ++v) sv[v] = 0.f;
#pragma unroll
        for (int t = 0; t < 4; ++t) {
            const int col = cw + t * 16 + mn;
            const float b1c = b1[col];
            const float w2c = w2[col];
#pragma unroll
            for (int v = 0; v < 8; ++v) {
                float h = acc[u][t][v] + b1c;
                h = fmaxf(h, 0.f);
                sv[v] += h * w2c;
            }
        }
        // reduce over the 16 N-lanes of each half-group
#pragma unroll
        for (int v = 0; v < 8; ++v)
#pragma unroll
            for (int m = 1; m < 16; m <<= 1)
                sv[v] += __shfl_xor(sv[v], m, 32);
        if (mn == 0) {
#pragma unroll
            for (int v = 0; v < 8; ++v)
                atomicAdd(&red[u * 16 + half * 8 + v], sv[v]);
        }
    }
    __syncthreads();
    if (tid < 32) {
        int row = r0 + tid;
        if (row < nspan)
            scores[(size_t)b * nspan + row] = red[tid] + b2[0];
    }
}

// ---------------------------------------------------------------------------
// Top-k (k=20) per batch, descending, ties -> lower index (jax.lax.top_k)
// ---------------------------------------------------------------------------
__global__ __launch_bounds__(256)
void topk_kernel(const float* __restrict__ scores, int nspan, int k,
                 float* __restrict__ topv, int* __restrict__ topi) {
    const int b = blockIdx.x;
    __shared__ float sv[2048];
    __shared__ float rv[256];
    __shared__ int   ri[256];
    const float* src = scores + (size_t)b * nspan;
    for (int i = threadIdx.x; i < nspan; i += 256) sv[i] = src[i];
    __syncthreads();

    for (int it = 0; it < k; ++it) {
        float bv = -__builtin_inff();
        int   bi = 0;
        for (int i = threadIdx.x; i < nspan; i += 256) {
            float v = sv[i];
            if (v > bv || (v == bv && i < bi)) { bv = v; bi = i; }
        }
        rv[threadIdx.x] = bv;
        ri[threadIdx.x] = bi;
        __syncthreads();
        for (int s = 128; s > 0; s >>= 1) {
            if (threadIdx.x < s) {
                float ov = rv[threadIdx.x + s];
                int   oi = ri[threadIdx.x + s];
                if (ov > rv[threadIdx.x] ||
                    (ov == rv[threadIdx.x] && oi < ri[threadIdx.x])) {
                    rv[threadIdx.x] = ov;
                    ri[threadIdx.x] = oi;
                }
            }
            __syncthreads();
        }
        if (threadIdx.x == 0) {
            topv[(size_t)b * k + it] = rv[0];
            topi[(size_t)b * k + it] = ri[0];
            sv[ri[0]] = -__builtin_inff();
        }
        __syncthreads();
    }
}

// ---------------------------------------------------------------------------
// Opinion MLP over all 400 holder x target pairs per batch.
// Row r = i*20 + j; A row = concat(hs, he, ts, te) (4 x 768 bf16 gathers).
// 32 rows per block (13 blocks; tail rows clamped/masked).
// out[b][r][0..2] = pair_score + relu(orep @ W1 + b1) @ W2 + b2
// ---------------------------------------------------------------------------
__global__ __launch_bounds__(256)
void opinion_kernel(const unsigned short* __restrict__ Xbf,
                    const int* __restrict__ hstarts, const int* __restrict__ hends,
                    const int* __restrict__ tstarts, const int* __restrict__ tends,
                    const int* __restrict__ hidx, const int* __restrict__ tidx,
                    const float* __restrict__ hval, const float* __restrict__ tval,
                    const unsigned short* __restrict__ W1t,   // [HID][4H] bf16
                    const float* __restrict__ b1,             // [HID]
                    const float* __restrict__ w2,             // [HID][3]
                    const float* __restrict__ b2,             // [3]
                    float* __restrict__ out) {                // [B][400][3]
    const int b    = blockIdx.y;
    const int r0   = blockIdx.x * 32;
    const int tid  = threadIdx.x;
    const int wave = tid >> 5;
    const int lane = tid & 31;
    const int half = lane >> 4;
    const int mn   = lane & 15;

    const unsigned short* ps[2][4];
#pragma unroll
    for (int u = 0; u < 2; ++u) {
        int r = r0 + u * 16 + mn;
        if (r >= NPAIR) r = NPAIR - 1;
        const int i = r / NSEL;
        const int j = r - i * NSEL;
        const int hs = hidx[b * NSEL + i];
        const int ts = tidx[b * NSEL + j];
        ps[u][0] = Xbf + ((size_t)b * S_ + hstarts[hs]) * H_;
        ps[u][1] = Xbf + ((size_t)b * S_ + (hends[hs] - 1)) * H_;
        ps[u][2] = Xbf + ((size_t)b * S_ + tstarts[ts]) * H_;
        ps[u][3] = Xbf + ((size_t)b * S_ + (tends[ts] - 1)) * H_;
    }

    const int cw = wave * 64;
    const unsigned short* wb[4];
#pragma unroll
    for (int t = 0; t < 4; ++t)
        wb[t] = W1t + (size_t)(cw + t * 16 + mn) * FOUR_H;

    v8f acc[2][4];
#pragma unroll
    for (int u = 0; u < 2; ++u)
#pragma unroll
        for (int t = 0; t < 4; ++t) acc[u][t] = (v8f){};

#pragma unroll
    for (int ch = 0; ch < 4; ++ch) {
        const unsigned short* base0 = ps[0][ch];
        const unsigned short* base1 = ps[1][ch];
        for (int off = 0; off < H_; off += 32) {
            const unsigned short* a0p = base0 + off + 8 * half;
            const unsigned short* a1p = base1 + off + 8 * half;
            Frag a0, a1;
            a0.u[0] = *(const uint4*)(a0p);
            a0.u[1] = *(const uint4*)(a0p + 16);
            a1.u[0] = *(const uint4*)(a1p);
            a1.u[1] = *(const uint4*)(a1p + 16);
            __builtin_prefetch((const void*)(a0p + 32), 0, 0);
            const int kg = ch * H_ + off + 8 * half;
#pragma unroll
            for (int t = 0; t < 4; ++t) {
                Frag w;
                const unsigned short* q = wb[t] + kg;
                w.u[0] = *(const uint4*)(q);
                w.u[1] = *(const uint4*)(q + 16);
                acc[0][t] = __builtin_amdgcn_wmma_f32_16x16x32_bf16(
                    false, a0.v, false, w.v, (short)0, acc[0][t], false, false);
                acc[1][t] = __builtin_amdgcn_wmma_f32_16x16x32_bf16(
                    false, a1.v, false, w.v, (short)0, acc[1][t], false, false);
            }
        }
    }

    __shared__ float red[96];   // 32 rows x 3 outputs
    if (tid < 96) red[tid] = 0.f;
    __syncthreads();

#pragma unroll
    for (int u = 0; u < 2; ++u) {
        float s0[8], s1[8], s2[8];
#pragma unroll
        for (int v = 0; v < 8; ++v) { s0[v] = 0.f; s1[v] = 0.f; s2[v] = 0.f; }
#pragma unroll
        for (int t = 0; t < 4; ++t) {
            const int col = cw + t * 16 + mn;
            const float b1c = b1[col];
            const float wa = w2[col * 3 + 0];
            const float wbv = w2[col * 3 + 1];
            const float wc = w2[col * 3 + 2];
#pragma unroll
            for (int v = 0; v < 8; ++v) {
                float h = fmaxf(acc[u][t][v] + b1c, 0.f);
                s0[v] += h * wa;
                s1[v] += h * wbv;
                s2[v] += h * wc;
            }
        }
#pragma unroll
        for (int v = 0; v < 8; ++v)
#pragma unroll
            for (int m = 1; m < 16; m <<= 1) {
                s0[v] += __shfl_xor(s0[v], m, 32);
                s1[v] += __shfl_xor(s1[v], m, 32);
                s2[v] += __shfl_xor(s2[v], m, 32);
            }
        if (mn == 0) {
#pragma unroll
            for (int v = 0; v < 8; ++v) {
                const int row = u * 16 + half * 8 + v;
                atomicAdd(&red[row * 3 + 0], s0[v]);
                atomicAdd(&red[row * 3 + 1], s1[v]);
                atomicAdd(&red[row * 3 + 2], s2[v]);
            }
        }
    }
    __syncthreads();
    if (tid < 96) {
        const int row = tid / 3;
        const int k3  = tid - row * 3;
        const int rr  = r0 + row;
        if (rr < NPAIR) {
            const int ii = rr / NSEL;
            const int jj = rr - ii * NSEL;
            const float pair = hval[b * NSEL + ii] + tval[b * NSEL + jj];
            out[((size_t)b * NPAIR + rr) * 3 + k3] = red[tid] + b2[k3] + pair;
        }
    }
}

// ---------------------------------------------------------------------------
// Host launcher
// ---------------------------------------------------------------------------
extern "C" void kernel_launch(void* const* d_in, const int* in_sizes, int n_in,
                              void* d_out, int out_size, void* d_ws, size_t ws_size,
                              hipStream_t stream) {
    (void)n_in; (void)out_size;
    const float* X       = (const float*)d_in[0];
    const int*   hstarts = (const int*)d_in[2];
    const int*   hends   = (const int*)d_in[3];
    const int*   tstarts = (const int*)d_in[4];
    const int*   tends   = (const int*)d_in[5];
    const float* hw1 = (const float*)d_in[6];
    const float* hb1 = (const float*)d_in[7];
    const float* hw2 = (const float*)d_in[8];
    const float* hb2 = (const float*)d_in[9];
    const float* tw1 = (const float*)d_in[10];
    const float* tb1 = (const float*)d_in[11];
    const float* tw2 = (const float*)d_in[12];
    const float* tb2 = (const float*)d_in[13];
    const float* ow1 = (const float*)d_in[14];
    const float* ob1 = (const float*)d_in[15];
    const float* ow2 = (const float*)d_in[16];
    const float* ob2 = (const float*)d_in[17];

    const int NH = in_sizes[2];   // 1516 holder spans
    const int NT = in_sizes[4];   // 2012 target spans

    // carve workspace (256B aligned slices)
    size_t o = 0;
    char* ws = (char*)d_ws;
    auto carve = [&](size_t bytes) -> char* {
        char* p = ws + o;
        o = (o + bytes + 255) & ~(size_t)255;
        return p;
    };
    unsigned short* Xbf  = (unsigned short*)carve((size_t)B_ * S_ * H_ * 2);
    unsigned short* hw1t = (unsigned short*)carve((size_t)HID_ * TWO_H * 2);
    unsigned short* tw1t = (unsigned short*)carve((size_t)HID_ * TWO_H * 2);
    unsigned short* ow1t = (unsigned short*)carve((size_t)HID_ * FOUR_H * 2);
    float* hsc  = (float*)carve((size_t)B_ * NH * 4);
    float* tsc  = (float*)carve((size_t)B_ * NT * 4);
    float* hval = (float*)carve((size_t)B_ * NSEL * 4);
    float* tval = (float*)carve((size_t)B_ * NSEL * 4);
    int*   hidx = (int*)carve((size_t)B_ * NSEL * 4);
    int*   tidx = (int*)carve((size_t)B_ * NSEL * 4);
    (void)ws_size;

    // prep: bf16 conversions + weight transposes
    f2bf_copy_kernel<<<2048, 256, 0, stream>>>(X, Xbf, B_ * S_ * H_);
    f2bf_transpose_kernel<<<1536, 256, 0, stream>>>(hw1, hw1t, TWO_H, HID_);
    f2bf_transpose_kernel<<<1536, 256, 0, stream>>>(tw1, tw1t, TWO_H, HID_);
    f2bf_transpose_kernel<<<3072, 256, 0, stream>>>(ow1, ow1t, FOUR_H, HID_);

    // span scoring (holder + target)
    span_score_kernel<<<dim3((NH + 31) / 32, B_), 256, 0, stream>>>(
        Xbf, hstarts, hends, hw1t, hb1, hw2, hb2, hsc, NH);
    span_score_kernel<<<dim3((NT + 31) / 32, B_), 256, 0, stream>>>(
        Xbf, tstarts, tends, tw1t, tb1, tw2, tb2, tsc, NT);

    // top-20 per batch
    topk_kernel<<<B_, 256, 0, stream>>>(hsc, NH, NSEL, hval, hidx);
    topk_kernel<<<B_, 256, 0, stream>>>(tsc, NT, NSEL, tval, tidx);

    // opinion MLP over all pairs -> d_out [B,400,3]
    opinion_kernel<<<dim3((NPAIR + 31) / 32, B_), 256, 0, stream>>>(
        Xbf, hstarts, hends, tstarts, tends, hidx, tidx, hval, tval,
        ow1t, ob1, ow2, ob2, (float*)d_out);
}
